// MPGESNEncoder_42657615184407
// MI455X (gfx1250) — matrix operand order repaired and checked by reading, not compiled, not run on hardware
//
#include <hip/hip_runtime.h>
#include <stdint.h>

#define B_   32
#define S_   1024
#define DIN_ 128
#define H_   512
#define NN_  1024
#define NE_  16384
#define NBLK_ 32
#define NTHR_ 256

typedef __attribute__((ext_vector_type(16))) __bf16        bf16x16;
typedef __attribute__((ext_vector_type(8)))  float         f32x8;
typedef __attribute__((ext_vector_type(4)))  unsigned int  u32x4;

union FragU { u32x4 q[2]; bf16x16 v; };
static_assert(sizeof(FragU) == 32, "frag size");

__device__ __forceinline__ unsigned short f2bf(float f) {
  unsigned u = __float_as_uint(f);
  if ((u & 0x7F800000u) == 0x7F800000u) return (unsigned short)(u >> 16); // inf/nan
  unsigned r = u + 0x7FFFu + ((u >> 16) & 1u);                            // RNE
  return (unsigned short)(r >> 16);
}

// ---- WMMA fragment loaders -------------------------------------------------
// A (16x32 bf16, row-major source, rows = M across lanes 0..15):
//   lane half h = lane>>4; elements e=0..7 -> K = k0 + 8h + e
//                          elements e=8..15 -> K = k0 + 16 + 8h + (e-8)
__device__ __forceinline__ bf16x16 load_a_frag(const unsigned short* base, int rowStride,
                                               int m0, int k0, int lane) {
  const unsigned short* p = base + (size_t)(m0 + (lane & 15)) * rowStride + k0 + ((lane >> 4) << 3);
  FragU f;
  f.q[0] = *(const u32x4*)(p);
  f.q[1] = *(const u32x4*)(p + 16);
  return f.v;
}

// B (32x16 bf16) pre-packed: uint index = ((ntile*KB + kb)*32 + lane)*8 + vg
__device__ __forceinline__ bf16x16 load_b_frag(const unsigned int* pk, int ntile, int kb,
                                               int KB, int lane) {
  const unsigned int* q = pk + (((size_t)ntile * KB + kb) * 32 + lane) * 8;
  FragU f;
  f.q[0] = *(const u32x4*)(q);
  f.q[1] = *(const u32x4*)(q + 4);
  return f.v;
}

__device__ __forceinline__ f32x8 gemm_tile(const unsigned short* aBase, int aStride, int m0,
                                           const unsigned int* bPk, int ntile, int KB,
                                           int lane, f32x8 acc) {
  for (int kb = 0; kb < KB; ++kb) {
    bf16x16 a = load_a_frag(aBase, aStride, m0, kb << 5, lane);
    bf16x16 b = load_b_frag(bPk, ntile, kb, KB, lane);
    acc = __builtin_amdgcn_wmma_f32_16x16x32_bf16(false, a, false, b, (short)0, acc, false, false);
  }
  return acc;
}

// ---- grid-wide barrier (monotonic counter, device scope) -------------------
__device__ __forceinline__ void grid_sync(unsigned int* bar, int nblk) {
  __syncthreads();
  __threadfence();
  if (threadIdx.x == 0) {
    unsigned int prev = __hip_atomic_fetch_add(bar, 1u, __ATOMIC_ACQ_REL, __HIP_MEMORY_SCOPE_AGENT);
    unsigned int target = ((prev + 1u + (unsigned)nblk - 1u) / (unsigned)nblk) * (unsigned)nblk;
    while (__hip_atomic_load(bar, __ATOMIC_ACQUIRE, __HIP_MEMORY_SCOPE_AGENT) < target)
      __builtin_amdgcn_s_sleep(1);
  }
  __syncthreads();
  __threadfence();
}

// ---- persistent scan kernel: one launch = one ESN layer, all timesteps ----
__global__ void __launch_bounds__(NTHR_) esn_scan(
    const unsigned short* __restrict__ xbf, int xTStride, int xRowStride, int KBinp,
    const unsigned int* __restrict__ wresPk, const unsigned int* __restrict__ wpPk,
    const unsigned int* __restrict__ wuPk,   const unsigned int* __restrict__ winPk,
    const float* __restrict__ biasRes, const float* __restrict__ bp, const float* __restrict__ bu,
    float* __restrict__ hF, unsigned short* __restrict__ hBf,
    float* __restrict__ Rb, float* __restrict__ Ib, float* __restrict__ hgT,
    unsigned short* __restrict__ aggrBf,
    const int* __restrict__ rowptr, const int* __restrict__ csr,
    float* __restrict__ outF, unsigned short* __restrict__ outBf,
    float leak, unsigned int* __restrict__ bar, int nblk) {
  const int lane = threadIdx.x & 31;
  const int wid  = (blockIdx.x * blockDim.x + threadIdx.x) >> 5;
  const int nW   = (gridDim.x * blockDim.x) >> 5;

  for (int t = 0; t < S_; ++t) {
    // ---------- Phase 1: [res | hg | inp] GEMMs (256 16x16 tiles) ----------
    for (int tt = wid; tt < 256; tt += nW) {
      const int mt = tt >> 7, nt = tt & 127;
      const int m0 = mt << 4;
      f32x8 acc;
#pragma unroll
      for (int i = 0; i < 8; ++i) acc[i] = 0.0f;

      if (nt < 32) {                       // res = h @ Wres^T
        acc = gemm_tile(hBf, H_, m0, wresPk, nt, H_ / 32, lane, acc);
        const int col = (nt << 4) + (lane & 15);
#pragma unroll
        for (int r = 0; r < 8; ++r) {
          const int m = m0 + r + ((lane >> 4) << 3);
          Rb[m * H_ + col] = acc[r];
        }
      } else if (nt < 96) {                // hg = h @ Wp^T + bp  (stored transposed)
        const int nl = nt - 32;
        acc = gemm_tile(hBf, H_, m0, wpPk, nl, H_ / 32, lane, acc);
        const int c = (nl << 4) + (lane & 15);
        const float bpv = bp[c];
#pragma unroll
        for (int r = 0; r < 8; ++r) {
          const int m = m0 + r + ((lane >> 4) << 3);
          hgT[c * B_ + m] = acc[r] + bpv;
        }
      } else {                             // inp = x_t @ Win^T
        const int nl = nt - 96;
        acc = gemm_tile(xbf + (size_t)t * xTStride, xRowStride, m0, winPk, nl, KBinp, lane, acc);
        const int col = (nl << 4) + (lane & 15);
#pragma unroll
        for (int r = 0; r < 8; ++r) {
          const int m = m0 + r + ((lane >> 4) << 3);
          Ib[m * H_ + col] = acc[r];
        }
      }
    }
    grid_sync(bar, nblk);

    // ---------- Phase 2: CSR gather (deterministic scatter-add) ------------
    // lane = batch; node index wave-uniform -> scalarized CSR reads,
    // coalesced 128B row reads of hgT.
    for (int n = wid; n < NN_; n += nW) {
      const int beg = rowptr[n], end = rowptr[n + 1];
      float s = 0.0f;
      for (int e = beg; e < end; ++e) {
        const int sc = csr[e];
        s += hgT[sc * B_ + lane];
      }
      aggrBf[lane * NN_ + n] = f2bf(s);
    }
    grid_sync(bar, nblk);

    // ---------- Phase 3: gc GEMM + fused tanh / leaky update ---------------
    for (int tt = wid; tt < 64; tt += nW) {
      const int mt = tt >> 5, nt = tt & 31;
      const int m0 = mt << 4;
      f32x8 acc;
#pragma unroll
      for (int i = 0; i < 8; ++i) acc[i] = 0.0f;
      acc = gemm_tile(aggrBf, NN_, m0, wuPk, nt, NN_ / 32, lane, acc);
      const int col = (nt << 4) + (lane & 15);
      const float bsum = bu[col] + biasRes[col];
#pragma unroll
      for (int r = 0; r < 8; ++r) {
        const int m   = m0 + r + ((lane >> 4) << 3);
        const int idx = m * H_ + col;
        const float v    = acc[r] + bsum + Rb[idx] + Ib[idx];
        const float th   = tanhf(v);
        const float hold = hF[idx];
        const float hnew = hold + leak * (th - hold);
        hF[idx]  = hnew;
        hBf[idx] = f2bf(hnew);
        const size_t yi = ((size_t)m * S_ + t) * H_ + col;
        if (outF) outF[yi] = hnew; else outBf[yi] = f2bf(hnew);
      }
    }
    grid_sync(bar, nblk);
  }
}

// ---- one-time-per-call prep kernels ---------------------------------------

// edge_index may be int64 (reference) or int32 (jax default demotion).
// int64 little-endian with values < 1024 => all odd 32-bit words are zero.
__global__ void detect_kernel(const unsigned int* w, int n, int* flag) {
  int i = blockIdx.x * blockDim.x + threadIdx.x;
  if (i < n && w[2 * i + 1] != 0u) atomicOr(flag, 1);
}

__device__ __forceinline__ int edge_word(const unsigned int* w, int j, int stride) {
  return (int)w[(size_t)j * stride];
}

__global__ void counts_kernel(const unsigned int* w, const int* flag, int* counts) {
  int e = blockIdx.x * blockDim.x + threadIdx.x;
  if (e >= NE_) return;
  int stride = (*flag) ? 1 : 2;
  int d = edge_word(w, NE_ + e, stride);
  atomicAdd(&counts[d], 1);
}

__global__ void prefix_kernel(const int* counts, int* rowptr) {
  if (blockIdx.x == 0 && threadIdx.x == 0) {
    int acc = 0;
    rowptr[0] = 0;
    for (int n = 0; n < NN_; ++n) { acc += counts[n]; rowptr[n + 1] = acc; }
  }
}

// stable fill: one thread per destination node scans edges in order
__global__ void fill_kernel(const unsigned int* w, const int* flag,
                            const int* rowptr, int* csr) {
  int n = blockIdx.x * blockDim.x + threadIdx.x;
  if (n >= NN_) return;
  int stride = (*flag) ? 1 : 2;
  int p = rowptr[n];
  for (int e = 0; e < NE_; ++e) {
    if (edge_word(w, NE_ + e, stride) == n)
      csr[p++] = edge_word(w, e, stride);
  }
}

// Pack W (N x K row-major, f32) into WMMA-B fragment order (bf16 pairs).
// B element e of lane: n = ntile*16 + (lane&15), k = kb*32 + 8*(lane>>4) + (e<8 ? e : 16 + (e-8)).
__global__ void pack_b_kernel(const float* __restrict__ W, unsigned int* __restrict__ out,
                              int N, int K) {
  int u = blockIdx.x * blockDim.x + threadIdx.x;
  int total = (N * K) >> 1;
  if (u >= total) return;
  int vg    = u & 7;
  int lane  = (u >> 3) & 31;
  int rest  = u >> 8;
  int KB    = K >> 5;
  int kb    = rest % KB;
  int ntile = rest / KB;
  int n = (ntile << 4) + (lane & 15);
  int k = (kb << 5) + ((vg & 3) << 1) + ((vg >> 2) << 4) + ((lane >> 4) << 3);
  float f0 = W[(size_t)n * K + k];
  float f1 = W[(size_t)n * K + k + 1];
  out[u] = (unsigned int)f2bf(f0) | ((unsigned int)f2bf(f1) << 16);
}

__global__ void cvt_bf16_kernel(const float* __restrict__ in, unsigned int* __restrict__ out,
                                int nPairs) {
  int i = blockIdx.x * blockDim.x + threadIdx.x;
  if (i >= nPairs) return;
  out[i] = (unsigned int)f2bf(in[2 * i]) | ((unsigned int)f2bf(in[2 * i + 1]) << 16);
}

// ---- host orchestration ----------------------------------------------------
extern "C" void kernel_launch(void* const* d_in, const int* in_sizes, int n_in,
                              void* d_out, int out_size, void* d_ws, size_t ws_size,
                              hipStream_t stream) {
  (void)in_sizes; (void)n_in; (void)out_size; (void)ws_size;

  const float*        x     = (const float*)d_in[0];
  const unsigned int* edges = (const unsigned int*)d_in[1];
  const float* Win0  = (const float*)d_in[2];
  const float* Wres0 = (const float*)d_in[3];
  const float* bias0 = (const float*)d_in[4];
  const float* Wp0   = (const float*)d_in[5];
  const float* bp0   = (const float*)d_in[6];
  const float* Wu0   = (const float*)d_in[7];
  const float* bu0   = (const float*)d_in[8];
  const float* Win1  = (const float*)d_in[9];
  const float* Wres1 = (const float*)d_in[10];
  const float* bias1 = (const float*)d_in[11];
  const float* Wp1   = (const float*)d_in[12];
  const float* bp1   = (const float*)d_in[13];
  const float* Wu1   = (const float*)d_in[14];
  const float* bu1   = (const float*)d_in[15];
  float* out = (float*)d_out;

  char* ws = (char*)d_ws;
  size_t off = 0;
  auto alloc = [&](size_t bytes) -> char* {
    off = (off + 255) & ~(size_t)255;
    char* p = ws + off;
    off += bytes;
    return p;
  };

  unsigned int*   bar     = (unsigned int*)alloc(4);
  int*            flag    = (int*)alloc(4);
  int*            counts  = (int*)alloc((size_t)NN_ * 4);
  int*            rowptr  = (int*)alloc((size_t)(NN_ + 1) * 4);
  int*            csr     = (int*)alloc((size_t)NE_ * 4);
  unsigned short* xbf     = (unsigned short*)alloc((size_t)B_ * S_ * DIN_ * 2);
  unsigned short* y0bf    = (unsigned short*)alloc((size_t)B_ * S_ * H_ * 2);
  float*          hF      = (float*)alloc((size_t)B_ * H_ * 4);
  unsigned short* hBf     = (unsigned short*)alloc((size_t)B_ * H_ * 2);
  float*          Rb      = (float*)alloc((size_t)B_ * H_ * 4);
  float*          Ib      = (float*)alloc((size_t)B_ * H_ * 4);
  float*          hgT     = (float*)alloc((size_t)NN_ * B_ * 4);
  unsigned short* aggrBf  = (unsigned short*)alloc((size_t)B_ * NN_ * 2);
  unsigned int*   wres0Pk = (unsigned int*)alloc((size_t)H_ * H_ * 2);
  unsigned int*   wp0Pk   = (unsigned int*)alloc((size_t)NN_ * H_ * 2);
  unsigned int*   wu0Pk   = (unsigned int*)alloc((size_t)H_ * NN_ * 2);
  unsigned int*   win0Pk  = (unsigned int*)alloc((size_t)H_ * DIN_ * 2);
  unsigned int*   wres1Pk = (unsigned int*)alloc((size_t)H_ * H_ * 2);
  unsigned int*   wp1Pk   = (unsigned int*)alloc((size_t)NN_ * H_ * 2);
  unsigned int*   wu1Pk   = (unsigned int*)alloc((size_t)H_ * NN_ * 2);
  unsigned int*   win1Pk  = (unsigned int*)alloc((size_t)H_ * H_ * 2);

  hipMemsetAsync(bar, 0, 4, stream);
  hipMemsetAsync(flag, 0, 4, stream);
  hipMemsetAsync(counts, 0, (size_t)NN_ * 4, stream);

  detect_kernel<<<32, 256, 0, stream>>>(edges, 8192, flag);
  counts_kernel<<<NE_ / 256, 256, 0, stream>>>(edges, flag, counts);
  prefix_kernel<<<1, 1, 0, stream>>>(counts, rowptr);
  fill_kernel<<<NN_ / 256, 256, 0, stream>>>(edges, flag, rowptr, csr);

  auto launch_pack = [&](const float* W, unsigned int* o, int N, int K) {
    int total = (N * K) >> 1;
    pack_b_kernel<<<(total + 255) / 256, 256, 0, stream>>>(W, o, N, K);
  };
  launch_pack(Wres0, wres0Pk, H_, H_);
  launch_pack(Wp0,   wp0Pk,   NN_, H_);
  launch_pack(Wu0,   wu0Pk,   H_, NN_);
  launch_pack(Win0,  win0Pk,  H_, DIN_);
  launch_pack(Wres1, wres1Pk, H_, H_);
  launch_pack(Wp1,   wp1Pk,   NN_, H_);
  launch_pack(Wu1,   wu1Pk,   H_, NN_);
  launch_pack(Win1,  win1Pk,  H_, H_);

  int nx = (B_ * S_ * DIN_) >> 1;
  cvt_bf16_kernel<<<(nx + 255) / 256, 256, 0, stream>>>(x, (unsigned int*)xbf, nx);

  // -------- layer 0: x -> y0 (bf16) ----------------------------------------
  hipMemsetAsync(hF, 0, (size_t)B_ * H_ * 4, stream);
  hipMemsetAsync(hBf, 0, (size_t)B_ * H_ * 2, stream);
  esn_scan<<<NBLK_, NTHR_, 0, stream>>>(
      xbf, DIN_, S_ * DIN_, DIN_ / 32,
      wres0Pk, wp0Pk, wu0Pk, win0Pk, bias0, bp0, bu0,
      hF, hBf, Rb, Ib, hgT, aggrBf, rowptr, csr,
      nullptr, y0bf, 0.1f, bar, NBLK_);

  // -------- layer 1: y0 -> out (f32) ---------------------------------------
  hipMemsetAsync(hF, 0, (size_t)B_ * H_ * 4, stream);
  hipMemsetAsync(hBf, 0, (size_t)B_ * H_ * 2, stream);
  esn_scan<<<NBLK_, NTHR_, 0, stream>>>(
      y0bf, H_, S_ * H_, H_ / 32,
      wres1Pk, wp1Pk, wu1Pk, win1Pk, bias1, bp1, bu1,
      hF, hBf, Rb, Ib, hgT, aggrBf, rowptr, csr,
      out, nullptr, 0.15f, bar, NBLK_);
}
